// ModelTorch_28681791602776
// MI455X (gfx1250) — matrix operand order repaired and checked by reading, not compiled, not run on hardware
//
#include <hip/hip_runtime.h>
#include <hip/hip_bf16.h>

// Problem constants (from reference setup)
#define B_  1
#define H_  16
#define HKV_ 4
#define S_  2048
#define D_  128
#define GQA_ 4

typedef __attribute__((ext_vector_type(16))) __bf16 v16bf;
typedef __attribute__((ext_vector_type(8)))  float  v8f;

// LDS row strides (in ushort elements); keep 8-byte row alignment for packed stores
#define KS 136   // K tile row stride   (32 rows x 128 d)
#define VS 40    // V^T tile row stride (128 rows x 32 keys)
#define PS 40    // P tile row stride   (16 rows x 32 keys)

__device__ __forceinline__ unsigned short f2bfu(float f) {
    // native hardware f32->bf16 convert (RNE)
    return __builtin_bit_cast(unsigned short, (__bf16)f);
}
__device__ __forceinline__ unsigned pk2(float a, float b) {
    return (unsigned)f2bfu(a) | ((unsigned)f2bfu(b) << 16);
}
__device__ __forceinline__ __bf16 u2bf(unsigned short u) {
    return __builtin_bit_cast(__bf16, u);
}

// xor-butterfly step within each 16-lane group, pure VALU (v_permlane16_b32)
#if __has_builtin(__builtin_amdgcn_permlane16)
__device__ __forceinline__ float plane16(float x, unsigned sel_lo, unsigned sel_hi) {
    unsigned u = __builtin_bit_cast(unsigned, x);
    unsigned r = __builtin_amdgcn_permlane16(u, u, sel_lo, sel_hi, false, false);
    return __builtin_bit_cast(float, r);
}
__device__ __forceinline__ float rowmax16(float x) {
    x = fmaxf(x, plane16(x, 0x67452301u, 0xEFCDAB89u));  // xor 1
    x = fmaxf(x, plane16(x, 0x54761032u, 0xDCFE98BAu));  // xor 2
    x = fmaxf(x, plane16(x, 0x32107654u, 0xBA98FEDCu));  // xor 4
    x = fmaxf(x, plane16(x, 0xFEDCBA98u, 0x76543210u));  // xor 8
    return x;
}
#else
__device__ __forceinline__ float rowmax16(float x) {
    #pragma unroll
    for (int m = 1; m < 16; m <<= 1) x = fmaxf(x, __shfl_xor(x, m, 32));
    return x;
}
#endif

__global__ __launch_bounds__(128) void fa_gqa_causal_kernel(
    const float* __restrict__ Qg,
    const float* __restrict__ Kg,
    const float* __restrict__ Vg,
    float* __restrict__ Og) {

    __shared__ unsigned short sK[32 * KS];        // [key][d]  bf16
    __shared__ unsigned short sVT[128 * VS];      // [d][key]  bf16 (transposed)
    __shared__ unsigned short sQ[4][16 * KS];     // per-wave  [m][d]
    __shared__ unsigned short sP[4][16 * PS];     // per-wave  [m][key]

    const int tid  = threadIdx.x;          // 0..127
    const int lane = tid & 31;
    // scalarize the wave id so per-wave control flow is SALU (EXEC stays all-1s)
    const int wave = __builtin_amdgcn_readfirstlane(tid >> 5);
    const int g    = lane >> 4;            // lane half (0/1)
    const int n    = lane & 15;            // column / M-sub index

    const int h     = blockIdx.y;          // 0..15
    const int hkv   = h / GQA_;
    const int qbase = blockIdx.x * 64;     // WG query base
    const int q0    = qbase + wave * 16;   // this wave's query rows q0..q0+15

    const float scale = 0.08838834764831845f;  // 1/sqrt(128)
    const float NEG_INF = -__builtin_inff();

    const float* Qh = Qg + (size_t)h   * S_ * D_;
    const float* Kh = Kg + (size_t)hkv * S_ * D_;
    const float* Vh = Vg + (size_t)hkv * S_ * D_;
    float*       Oh = Og + (size_t)h   * S_ * D_;

    // ---- Stage this wave's 16x128 Q tile into LDS as bf16 (float4 + packed b64) ----
    for (int i = lane * 4; i < 16 * 128; i += 32 * 4) {
        int r = i >> 7, c = i & 127;
        const float4 qf = *(const float4*)&Qh[(size_t)(q0 + r) * D_ + c];
        unsigned* dst = (unsigned*)&sQ[wave][r * KS + c];
        dst[0] = pk2(qf.x, qf.y);
        dst[1] = pk2(qf.z, qf.w);
    }
    __syncthreads();

    // ---- Build Q A-matrix registers: 4 chunks of 16x32 (contiguous k order, matching B) ----
    // lane l holds M = l%16; half-index j -> k = 16g+j
    v16bf aQ[4];
    #pragma unroll
    for (int d0 = 0; d0 < 4; ++d0) {
        #pragma unroll
        for (int j = 0; j < 16; ++j) {
            int k = 16 * g + j;
            aQ[d0][j] = u2bf(sQ[wave][n * KS + 32 * d0 + k]);
        }
    }

    // all-ones B matrix for WMMA row-sum: C[m][n] = sum_k A[m][k]
    v16bf onesB;
    #pragma unroll
    for (int j = 0; j < 16; ++j) onesB[j] = u2bf(0x3F80);   // bf16(1.0)

    // ---- Flash-attention state ----
    v8f acc[8];
    #pragma unroll
    for (int t = 0; t < 8; ++t) acc[t] = v8f{};
    float mrow[8], lrow[8];
    #pragma unroll
    for (int r = 0; r < 8; ++r) { mrow[r] = NEG_INF; lrow[r] = 0.0f; }

    const int nkb = (qbase + 64) >> 5;     // causal: key blocks of 32

    for (int kb = 0; kb < nkb; ++kb) {
        const int k0 = kb * 32;
        __syncthreads();   // previous iteration's sK/sVT readers done

        // ---- Stage K (row-major, packed) and V (transposed) tiles, fp32 -> bf16 ----
        for (int i = tid * 4; i < 32 * 128; i += 128 * 4) {
            int kr = i >> 7, c = i & 127;
            const float4 kf = *(const float4*)&Kh[(size_t)(k0 + kr) * D_ + c];
            unsigned* dk = (unsigned*)&sK[kr * KS + c];
            dk[0] = pk2(kf.x, kf.y);
            dk[1] = pk2(kf.z, kf.w);
            const float4 vf = *(const float4*)&Vh[(size_t)(k0 + kr) * D_ + c];
            sVT[(c + 0) * VS + kr] = f2bfu(vf.x);
            sVT[(c + 1) * VS + kr] = f2bfu(vf.y);
            sVT[(c + 2) * VS + kr] = f2bfu(vf.z);
            sVT[(c + 3) * VS + kr] = f2bfu(vf.w);
        }
        // Prefetch next tile (overlaps with compute; K/V are L2-resident anyway)
        if (kb + 1 < nkb) {
            const size_t off = (size_t)(k0 + 32 + (tid >> 2)) * D_ + (tid & 3) * 32;
            __builtin_prefetch(&Kh[off], 0, 1);
            __builtin_prefetch(&Vh[off], 0, 1);
        }
        __syncthreads();

        // Wave-uniform (SGPR) causal skip: this wave's rows are q0..q0+15
        const bool active = (k0 <= q0 + 15);
        float alphas[8];

        if (active) {
            // ---- S = Q K^T for two 16-key N-tiles ----
            v8f s0 = v8f{}, s1 = v8f{};
            #pragma unroll
            for (int d0 = 0; d0 < 4; ++d0) {
                v16bf bK0, bK1;
                #pragma unroll
                for (int j = 0; j < 16; ++j) {
                    // B layout: lane l holds B[16g + j][n]; B[k][n] = K[key n][32*d0 + k]
                    bK0[j] = u2bf(sK[n        * KS + 32 * d0 + 16 * g + j]);
                    bK1[j] = u2bf(sK[(16 + n) * KS + 32 * d0 + 16 * g + j]);
                }
                s0 = __builtin_amdgcn_wmma_f32_16x16x32_bf16(false, aQ[d0], false, bK0,
                                                             (short)0, s0, false, false);
                s1 = __builtin_amdgcn_wmma_f32_16x16x32_bf16(false, aQ[d0], false, bK1,
                                                             (short)0, s1, false, false);
            }

            // ---- Online softmax (C layout: VGPR r -> row q0 + r + 8g, col k0 + n) ----
            #pragma unroll
            for (int r = 0; r < 8; ++r) {
                const int qrow = q0 + r + 8 * g;
                const int key0 = k0 + n;
                const int key1 = k0 + 16 + n;
                float v0 = s0[r] * scale; if (key0 > qrow) v0 = NEG_INF;
                float v1 = s1[r] * scale; if (key1 > qrow) v1 = NEG_INF;

                const float smax = rowmax16(fmaxf(v0, v1));
                const float mnew = fmaxf(mrow[r], smax);
                const float alpha = __expf(mrow[r] - mnew);
                const float p0 = __expf(v0 - mnew);
                const float p1 = __expf(v1 - mnew);

                alphas[r] = alpha;
                mrow[r]   = mnew;
                #pragma unroll
                for (int t = 0; t < 8; ++t) acc[t][r] *= alpha;

                // store P in plain [m][key] order for A-layout reload
                sP[wave][(r + 8 * g) * PS + n]      = f2bfu(p0);
                sP[wave][(r + 8 * g) * PS + 16 + n] = f2bfu(p1);
            }
        }
        __syncthreads();   // WG-uniform barrier (sP visibility within each wave)

        if (active) {
            // ---- Reload P as A-matrix (16x32) ----
            v16bf aP;
            #pragma unroll
            for (int j = 0; j < 16; ++j) {
                int k = 16 * g + j;
                aP[j] = u2bf(sP[wave][n * PS + k]);
            }

            // ---- Row sums via WMMA against all-ones B: sums[r] = sum_k P[m_r][k] ----
            v8f sums = __builtin_amdgcn_wmma_f32_16x16x32_bf16(false, aP, false, onesB,
                                                               (short)0, v8f{}, false, false);
            #pragma unroll
            for (int r = 0; r < 8; ++r)
                lrow[r] = lrow[r] * alphas[r] + sums[r];

            // ---- PV: V^T as B (32x16 per d-tile) ----
            #pragma unroll
            for (int t = 0; t < 8; ++t) {
                v16bf bV;
                #pragma unroll
                for (int j = 0; j < 16; ++j)
                    bV[j] = u2bf(sVT[(16 * t + n) * VS + 16 * g + j]);
                acc[t] = __builtin_amdgcn_wmma_f32_16x16x32_bf16(false, aP, false, bV,
                                                                 (short)0, acc[t], false, false);
            }
        }
    }

    // ---- Finalize: O = acc / l ----
    #pragma unroll
    for (int r = 0; r < 8; ++r) {
        const int qrow = q0 + r + 8 * g;
        const float inv = 1.0f / lrow[r];
        #pragma unroll
        for (int t = 0; t < 8; ++t) {
            Oh[(size_t)qrow * D_ + 16 * t + n] = acc[t][r] * inv;
        }
    }
}

extern "C" void kernel_launch(void* const* d_in, const int* in_sizes, int n_in,
                              void* d_out, int out_size, void* d_ws, size_t ws_size,
                              hipStream_t stream) {
    (void)in_sizes; (void)n_in; (void)d_ws; (void)ws_size; (void)out_size;
    const float* Q = (const float*)d_in[0];
    const float* K = (const float*)d_in[1];
    const float* V = (const float*)d_in[2];
    float* O = (float*)d_out;
    dim3 grid(S_ / 64, H_);
    fa_gqa_causal_kernel<<<grid, 128, 0, stream>>>(Q, K, V, O);
}
